// KiperwasserDependencyParser_71992241816192
// MI455X (gfx1250) — compile-verified, matrix operand order checked by reading, not run on hardware
//
#include <hip/hip_runtime.h>
#include <hip/hip_bf16.h>

typedef __attribute__((ext_vector_type(16))) __bf16 v16bf;
typedef __attribute__((ext_vector_type(8)))  __bf16 v8bf;
typedef __attribute__((ext_vector_type(8)))  float  v8f;

#define H_   256
#define NTOK 1024

// ---------------------------------------------------------------- helpers
__device__ inline float fsigmoid(float x) { return 1.0f / (1.0f + __expf(-x)); }
__device__ inline float ftanh(float x) {
  float ax = fabsf(x);
  float e  = __expf(-2.0f * ax);
  float r  = (1.0f - e) / (1.0f + e);
  return copysignf(r, x);
}

__device__ inline v8f wmma_bf16(v16bf a, v16bf b, v8f c) {
  return __builtin_amdgcn_wmma_f32_16x16x32_bf16(
      /*neg_a=*/false, a, /*neg_b=*/false, b,
      /*c_mod=*/(short)0, c, /*reuse_a=*/false, /*reuse_b=*/false);
}

// Load a 16x32 bf16 A-fragment (or B-fragment from N x K weights) from
// row-major memory per CDNA5 ISA layout:
//   lane L holds row (L&15), K = {8*(L>>4)..+7} and {16+8*(L>>4)..+7}.
__device__ inline v16bf load_frag(const __bf16* __restrict__ base, int ld,
                                  int row, int k0, int lane) {
  const int r = lane & 15, half = lane >> 4;
  const __bf16* p = base + (size_t)(row + r) * ld + k0 + half * 8;
  v8bf lo = *(const v8bf*)p;          // K = k0 + 8*half .. +7
  v8bf hi = *(const v8bf*)(p + 16);   // K = k0 + 16 + 8*half .. +7
  v16bf f;
#pragma unroll
  for (int i = 0; i < 8; ++i) { f[i] = lo[i]; f[i + 8] = hi[i]; }
  return f;
}

// ---------------------------------------------------------------- convert/pad
// dst is dst_rows x dst_cols bf16, zero-padded outside src_rows x src_cols.
__global__ void k_cvt(const float* __restrict__ src, __bf16* __restrict__ dst,
                      int src_rows, int src_cols, int dst_cols, int total) {
  int idx = blockIdx.x * blockDim.x + threadIdx.x;
  if (idx >= total) return;
  int r = idx / dst_cols, c = idx - r * dst_cols;
  float v = (r < src_rows && c < src_cols) ? src[(size_t)r * src_cols + c] : 0.0f;
  dst[idx] = (__bf16)v;
}

// ---------------------------------------------------------------- embeddings
// X[i, 0:100) = word_emb[word_idx[i]], X[i,100:125) = pos_emb[pos_idx[i]], pad->128
__global__ void k_embed(const int* __restrict__ wi, const int* __restrict__ pi,
                        const float* __restrict__ we, const float* __restrict__ pe,
                        __bf16* __restrict__ X) {
  const int i = blockIdx.x, t = threadIdx.x;
  float v = 0.0f;
  if (t < 100)      v = we[(size_t)wi[i] * 100 + t];
  else if (t < 125) v = pe[(size_t)pi[i] * 25 + (t - 100)];
  X[(size_t)i * 128 + t] = (__bf16)v;
}

// ---------------------------------------------------------------- WMMA GEMM
// C[M x N] f32 = A[M x K] bf16 (row-major, lda) * W[N x K]^T bf16 (row-major, ldw)
//               (+ bias[n]).  Block = 8 waves -> 128 x 64 tile; wave -> 32 x 32.
__global__ __launch_bounds__(256) void k_gemm(
    const __bf16* __restrict__ A, int lda,
    const __bf16* __restrict__ W, int ldw,
    const float* __restrict__ bias,
    float* __restrict__ C, int ldc, int K) {
  const int lane = threadIdx.x & 31;
  const int wid  = threadIdx.x >> 5;
  const int wm = wid & 3, wn = wid >> 2;
  const int m0 = blockIdx.x * 128 + wm * 32;
  const int n0 = blockIdx.y * 64  + wn * 32;

  v8f acc00 = {}, acc01 = {}, acc10 = {}, acc11 = {};

  for (int k0 = 0; k0 < K; k0 += 32) {
    if (k0 + 32 < K) {  // prefetch next K-slab -> global_prefetch_b8
      __builtin_prefetch(A + (size_t)(m0 + (lane & 15)) * lda + k0 + 32, 0, 0);
      __builtin_prefetch(W + (size_t)(n0 + (lane & 15)) * ldw + k0 + 32, 0, 0);
    }
    v16bf a0 = load_frag(A, lda, m0,      k0, lane);
    v16bf a1 = load_frag(A, lda, m0 + 16, k0, lane);
    v16bf b0 = load_frag(W, ldw, n0,      k0, lane);
    v16bf b1 = load_frag(W, ldw, n0 + 16, k0, lane);
    acc00 = wmma_bf16(a0, b0, acc00);
    acc01 = wmma_bf16(a0, b1, acc01);
    acc10 = wmma_bf16(a1, b0, acc10);
    acc11 = wmma_bf16(a1, b1, acc11);
  }

  const int r = lane & 15, half = lane >> 4;
#pragma unroll
  for (int fm = 0; fm < 2; ++fm) {
#pragma unroll
    for (int fn = 0; fn < 2; ++fn) {
      v8f acc = (fm == 0) ? ((fn == 0) ? acc00 : acc01)
                          : ((fn == 0) ? acc10 : acc11);
      const int col  = n0 + fn * 16 + r;
      const int rowb = m0 + fm * 16 + half * 8;
      const float bv = bias ? bias[col] : 0.0f;
#pragma unroll
      for (int i = 0; i < 8; ++i)
        C[(size_t)(rowb + i) * ldc + col] = acc[i] + bv;
    }
  }
}

// ---------------------------------------------------------------- LSTM recurrence
// One block per direction (blockIdx.x = dir; dir 1 scans in reverse).
// Z is precomputed Wih@x_t + b, shape [2][1024][1024] f32.
// Whh is the ORIGINAL f32 weights [2][1024][256] (L2-resident).
// Writes bf16 hidden states into Hb[t][dir*256 + j] (1024 x 512 concat).
__global__ __launch_bounds__(1024) void k_lstm(
    const float* __restrict__ Z, const float* __restrict__ Whh,
    __bf16* __restrict__ Hb) {
  const int dir = blockIdx.x;
  const int tid = threadIdx.x;
  __shared__ __align__(16) float h_s[H_];
  __shared__ float zb[4 * H_];
  if (tid < H_) h_s[tid] = 0.0f;
  float c = 0.0f;
  const float4* Wr = (const float4*)(Whh + (size_t)dir * 4 * H_ * H_ + (size_t)tid * H_);
  const float*  Zd = Z + (size_t)dir * NTOK * 4 * H_;
  __syncthreads();

  for (int s = 0; s < NTOK; ++s) {
    const int t = dir ? (NTOK - 1 - s) : s;
    float acc = Zd[(size_t)t * 4 * H_ + tid];
    const float4* hv = (const float4*)h_s;
#pragma unroll 8
    for (int k = 0; k < H_ / 4; ++k) {
      float4 w = Wr[k]; float4 h4 = hv[k];
      acc += w.x * h4.x + w.y * h4.y + w.z * h4.z + w.w * h4.w;
    }
    zb[tid] = acc;
    __syncthreads();  // all reads of h_s done, all zb written
    if (tid < H_) {
      const float ig = fsigmoid(zb[tid]);
      const float fg = fsigmoid(zb[H_ + tid]);
      const float gg = ftanh(zb[2 * H_ + tid]);
      const float og = fsigmoid(zb[3 * H_ + tid]);
      c = fg * c + ig * gg;
      const float h = og * ftanh(c);
      h_s[tid] = h;
      Hb[(size_t)t * (2 * H_) + dir * H_ + tid] = (__bf16)h;
    }
    __syncthreads();  // h_s updated before next step's reads
  }
}

// ---------------------------------------------------------------- biaffine scores
// score[i,j] = sum_h W2[h]*tanh(a[i,h] + b[j,h] + b1[h]) + b2   (h < 100)
// a,b strides are 128 (padded GEMM output).  16x16 tile per 256-thread block.
__global__ __launch_bounds__(256) void k_biaffine(
    const float* __restrict__ Aa, const float* __restrict__ Bb,
    const float* __restrict__ b1, const float* __restrict__ W2,
    const float* __restrict__ b2, float* __restrict__ out) {
  __shared__ float aS[16 * 100];
  __shared__ float bS[16 * 100];
  __shared__ float w2s[100];
  const int tid = threadIdx.x;
  const int i0 = blockIdx.x * 16, j0 = blockIdx.y * 16;
  for (int idx = tid; idx < 16 * 100; idx += 256) {
    int r = idx / 100, h = idx - r * 100;
    aS[idx] = Aa[(size_t)(i0 + r) * 128 + h] + b1[h];  // fold bias into a
    bS[idx] = Bb[(size_t)(j0 + r) * 128 + h];
  }
  if (tid < 100) w2s[tid] = W2[tid];
  __syncthreads();
  const int ti = tid >> 4, tj = tid & 15;
  const float* ar = aS + ti * 100;
  const float* br = bS + tj * 100;
  float s = 0.0f;
#pragma unroll 4
  for (int h = 0; h < 100; ++h) s += w2s[h] * ftanh(ar[h] + br[h]);
  out[(size_t)(i0 + ti) * NTOK + (j0 + tj)] = s + b2[0];
}

// ---------------------------------------------------------------- launch
extern "C" void kernel_launch(void* const* d_in, const int* in_sizes, int n_in,
                              void* d_out, int out_size, void* d_ws, size_t ws_size,
                              hipStream_t stream) {
  (void)in_sizes; (void)n_in; (void)out_size; (void)ws_size;
  const int*   word_idx = (const int*)  d_in[0];
  const int*   pos_idx  = (const int*)  d_in[1];
  const float* word_emb = (const float*)d_in[2];
  const float* pos_emb  = (const float*)d_in[3];
  const float* wih0     = (const float*)d_in[4];
  const float* whh0     = (const float*)d_in[5];
  const float* b0       = (const float*)d_in[6];
  const float* wih1     = (const float*)d_in[7];
  const float* whh1     = (const float*)d_in[8];
  const float* b1       = (const float*)d_in[9];
  const float* mW1      = (const float*)d_in[10];
  const float* mb1      = (const float*)d_in[11];
  const float* mW2      = (const float*)d_in[12];
  const float* mb2      = (const float*)d_in[13];
  float* out = (float*)d_out;

  char* ws = (char*)d_ws;                              // bytes      cum
  __bf16* Xb    = (__bf16*)(ws);                       // 256 KB  @ 0
  __bf16* Wih0b = (__bf16*)(ws + ( 256u << 10));       // 512 KB  @ 256K
  __bf16* Wih1b = (__bf16*)(ws + ( 768u << 10));       // 2 MB    @ 768K
  __bf16* Wmlpb = (__bf16*)(ws + (2816u << 10));       // 256 KB  @ 2.75M
  __bf16* H0b   = (__bf16*)(ws + (3072u << 10));       // 1 MB    @ 3M
  __bf16* H1b   = (__bf16*)(ws + (4096u << 10));       // 1 MB    @ 4M
  float*  Z0    = (float*) (ws + (5120u << 10));       // 8 MB    @ 5M
  float*  Z1    = (float*) (ws + (13312u << 10));      // 8 MB    @ 13M
  float*  Am    = (float*) (ws + (21504u << 10));      // 512 KB  @ 21M
  float*  Bm    = (float*) (ws + (22016u << 10));      // 512 KB  @ 21.5M

  // 1) embeddings + bf16 weight staging (padded for WMMA alignment)
  k_embed<<<NTOK, 128, 0, stream>>>(word_idx, pos_idx, word_emb, pos_emb, Xb);
  { int tot = 2048 * 128;
    k_cvt<<<(tot + 255) / 256, 256, 0, stream>>>(wih0, Wih0b, 2048, 125, 128, tot); }
  { int tot = 2048 * 512;
    k_cvt<<<(tot + 255) / 256, 256, 0, stream>>>(wih1, Wih1b, 2048, 512, 512, tot); }
  { int tot = 128 * 1024;  // pad 100 rows -> 128
    k_cvt<<<(tot + 255) / 256, 256, 0, stream>>>(mW1, Wmlpb, 100, 1024, 1024, tot); }

  // 2) layer-0 input projections (WMMA), then recurrence
  for (int d = 0; d < 2; ++d)
    k_gemm<<<dim3(8, 16), 256, 0, stream>>>(
        Xb, 128, Wih0b + (size_t)d * 1024 * 128, 128,
        b0 + (size_t)d * 1024, Z0 + (size_t)d * 1024 * 1024, 1024, 128);
  k_lstm<<<2, 1024, 0, stream>>>(Z0, whh0, H0b);

  // 3) layer-1 input projections (WMMA), then recurrence
  for (int d = 0; d < 2; ++d)
    k_gemm<<<dim3(8, 16), 256, 0, stream>>>(
        H0b, 512, Wih1b + (size_t)d * 1024 * 512, 512,
        b1 + (size_t)d * 1024, Z1 + (size_t)d * 1024 * 1024, 1024, 512);
  k_lstm<<<2, 1024, 0, stream>>>(Z1, whh1, H1b);

  // 4) MLP projections a = H1@Wa^T, b = H1@Wb^T (WMMA), padded N=128
  k_gemm<<<dim3(8, 2), 256, 0, stream>>>(H1b, 512, Wmlpb,       1024,
                                         (const float*)nullptr, Am, 128, 512);
  k_gemm<<<dim3(8, 2), 256, 0, stream>>>(H1b, 512, Wmlpb + 512, 1024,
                                         (const float*)nullptr, Bm, 128, 512);

  // 5) biaffine tanh-dot over 100 hidden units -> 1024x1024 scores
  k_biaffine<<<dim3(64, 64), 256, 0, stream>>>(Am, Bm, mb1, mW2, mb2, out);
}
